// DifferentiableRankIntegration_16819091931306
// MI455X (gfx1250) — matrix operand order repaired and checked by reading, not compiled, not run on hardware
//
#include <hip/hip_runtime.h>

typedef _Float16 v16h __attribute__((ext_vector_type(16)));
typedef _Float16 v2h  __attribute__((ext_vector_type(2)));
typedef __fp16   v2hf __attribute__((ext_vector_type(2)));
typedef float    v8f  __attribute__((ext_vector_type(8)));

#define B_DIM    1024
#define NTHREADS 256
#define NWAVES   8

// sigmoid(x) = 0.5 + 0.5*tanh(x/2); gfx1250 has hardware v_tanh_f32.
__device__ __forceinline__ float tanh_hw(float x) {
#if __has_builtin(__builtin_amdgcn_tanhf)
  return __builtin_amdgcn_tanhf(x);
#else
  float r;
  asm volatile("v_tanh_f32 %0, %1" : "=v"(r) : "v"(x));
  return r;
#endif
}

__device__ __forceinline__ v2h pk16(float a, float b) {
#if __has_builtin(__builtin_amdgcn_cvt_pkrtz)
  union { v2hf f; v2h h; } u;
  u.f = __builtin_amdgcn_cvt_pkrtz(a, b);
  return u.h;
#else
  v2h r; r.x = (_Float16)a; r.y = (_Float16)b; return r;
#endif
}

union AU { v2h  p[8]; v16h h; };  // 16 f16 tanh values (A operand)
union BU { int4 q[2]; v16h h; };  // 16 f16 weights     (B operand)

// Accumulate D[m=b, n] += sum_a tanh(5*(s[a]-s[b])) * W[a,n] over a = 0..1023,
// K=32 per WMMA. Per ISA layout (16-bit A 16x32): lanes 0-15 hold K {0..7,16..23},
// lanes 16-31 hold K {8..15,24..31}  -> base offset kh*8, chunks +0 and +16.
__device__ __forceinline__ v8f rank_wmma(const float* __restrict__ srow,
                                         const _Float16* __restrict__ wbase,
                                         int wstep, int kh, float nsb5) {
  v8f acc = {0.f, 0.f, 0.f, 0.f, 0.f, 0.f, 0.f, 0.f};
#pragma unroll 4
  for (int k = 0; k < 32; ++k) {
    const float* sa = srow + k * 32 + kh * 8;
    float4 s0 = *(const float4*)(sa + 0);
    float4 s1 = *(const float4*)(sa + 4);
    float4 s2 = *(const float4*)(sa + 16);
    float4 s3 = *(const float4*)(sa + 20);
    AU A;
    A.p[0] = pk16(tanh_hw(fmaf(s0.x, 5.f, nsb5)), tanh_hw(fmaf(s0.y, 5.f, nsb5)));
    A.p[1] = pk16(tanh_hw(fmaf(s0.z, 5.f, nsb5)), tanh_hw(fmaf(s0.w, 5.f, nsb5)));
    A.p[2] = pk16(tanh_hw(fmaf(s1.x, 5.f, nsb5)), tanh_hw(fmaf(s1.y, 5.f, nsb5)));
    A.p[3] = pk16(tanh_hw(fmaf(s1.z, 5.f, nsb5)), tanh_hw(fmaf(s1.w, 5.f, nsb5)));
    A.p[4] = pk16(tanh_hw(fmaf(s2.x, 5.f, nsb5)), tanh_hw(fmaf(s2.y, 5.f, nsb5)));
    A.p[5] = pk16(tanh_hw(fmaf(s2.z, 5.f, nsb5)), tanh_hw(fmaf(s2.w, 5.f, nsb5)));
    A.p[6] = pk16(tanh_hw(fmaf(s3.x, 5.f, nsb5)), tanh_hw(fmaf(s3.y, 5.f, nsb5)));
    A.p[7] = pk16(tanh_hw(fmaf(s3.z, 5.f, nsb5)), tanh_hw(fmaf(s3.w, 5.f, nsb5)));
    BU Bm;
    const int4* wp = (const int4*)(wbase + k * wstep);
    Bm.q[0] = wp[0];
    Bm.q[1] = wp[2];
    acc = __builtin_amdgcn_wmma_f32_16x16x32_f16(false, A.h, false, Bm.h,
                                                 (short)0, acc, false, false);
  }
  return acc;
}

__global__ __launch_bounds__(NTHREADS)
void soft_rank_fused_kernel(const float* __restrict__ g_sv,
                            const float* __restrict__ g_sl,
                            const unsigned char* __restrict__ g_pos,
                            const unsigned char* __restrict__ g_neg,
                            const float* __restrict__ g_wv,
                            const float* __restrict__ g_wl,
                            float* __restrict__ g_out) {
  __shared__ __align__(16) float    sv[B_DIM];
  __shared__ __align__(16) float    sl[B_DIM];
  __shared__ __align__(16) _Float16 nfh[B_DIM];
  __shared__ __align__(16) _Float16 pfh[B_DIM];
  __shared__ __align__(16) _Float16 zeroH[32];
  __shared__ float stage[NWAVES][4][16];  // {NF_v, PF_v, NF_l, PF_l}
  __shared__ int   cnt[2];                // {Npos, Nneg}

  const int  tid  = threadIdx.x;
  const int  lane = tid & 31;
  const int  wave = tid >> 5;
  const int  sel  = lane & 15;
  const int  kh   = lane >> 4;
  const int  c    = blockIdx.x;
  const long row  = (long)c * B_DIM;

  if (tid < 2)  cnt[tid] = 0;
  if (tid < 32) zeroH[tid] = (_Float16)0.f;
  {
    int i = tid * 4;
    *(float4*)&sv[i] = *(const float4*)&g_sv[row + i];
    *(float4*)&sl[i] = *(const float4*)&g_sl[row + i];
  }
  __syncthreads();
  int np = 0, nn = 0;
#pragma unroll
  for (int j = 0; j < 4; ++j) {
    int i = tid * 4 + j;
    unsigned char p = g_pos[row + i];
    unsigned char n = g_neg[row + i];
    pfh[i] = (_Float16)(float)p;
    nfh[i] = (_Float16)(float)n;
    np += p;
    nn += n;
  }
  atomicAdd(&cnt[0], np);
  atomicAdd(&cnt[1], nn);
  __syncthreads();
  const float fNpos = (float)cnt[0];
  const float fNneg = (float)cnt[1];

  // B operand: column N=0 carries nf, N=1 carries pf, all other columns zero.
  // Per 16-bit B 32x16 layout: lanes 0-15 hold K {0..7,16..23}, lanes 16-31 K {8..15,24..31}.
  const _Float16* wbase = zeroH;
  int wstep = 0;
  if (sel == 0)      { wbase = nfh + kh * 8; wstep = 32; }
  else if (sel == 1) { wbase = pfh + kh * 8; wstep = 32; }

  for (int t = 0; t < 8; ++t) {
    const int b0 = (wave * 8 + t) * 16;
    const float nsb5_v = -5.f * sv[b0 + sel];
    const float nsb5_l = -5.f * sl[b0 + sel];
    v8f av = rank_wmma(sv, wbase, wstep, kh, nsb5_v);
    v8f al = rank_wmma(sl, wbase, wstep, kh, nsb5_l);
    __syncthreads();
    if (sel < 2) {
#pragma unroll
      for (int i = 0; i < 8; ++i) {
        stage[wave][sel][kh * 8 + i]     = av[i];
        stage[wave][2 + sel][kh * 8 + i] = al[i];
      }
    }
    __syncthreads();
    if (lane < 16) {
      int b = b0 + lane;
      float thNFv = stage[wave][0][lane];
      float thPFv = stage[wave][1][lane];
      float thNFl = stage[wave][2][lane];
      float thPFl = stage[wave][3][lane];
      float pfb = (float)g_pos[row + b];
      float nfb = (float)g_neg[row + b];
      // rank_pos = 1 + sum sig*nf = 1 + 0.5*(Nneg + sum tanh*nf), etc.
      float rv = (1.f + 0.5f * (fNneg + thNFv)) * pfb
               + (1.f + 0.5f * (fNpos + thPFv)) * nfb;
      float rl = (1.f + 0.5f * (fNneg + thNFl)) * pfb
               + (1.f + 0.5f * (fNpos + thPFl)) * nfb;
      float o = 61.f * (g_wv[row + b] * __builtin_amdgcn_rcpf(60.f + rv) +
                        g_wl[row + b] * __builtin_amdgcn_rcpf(60.f + rl));
      g_out[row + b] = o;
    }
  }
}

extern "C" void kernel_launch(void* const* d_in, const int* in_sizes, int n_in,
                              void* d_out, int out_size, void* d_ws, size_t ws_size,
                              hipStream_t stream) {
  (void)in_sizes; (void)n_in; (void)out_size; (void)d_ws; (void)ws_size;
  const float*         s_v = (const float*)d_in[0];
  const float*         s_l = (const float*)d_in[1];
  const unsigned char* pos = (const unsigned char*)d_in[2];  // jax bool_: 1 byte/elem
  const unsigned char* neg = (const unsigned char*)d_in[3];
  const float*         w_v = (const float*)d_in[4];
  const float*         w_l = (const float*)d_in[5];
  float*               out = (float*)d_out;
  soft_rank_fused_kernel<<<B_DIM, NTHREADS, 0, stream>>>(s_v, s_l, pos, neg,
                                                         w_v, w_l, out);
}
